// MambaEncoder_59734405153282
// MI455X (gfx1250) — compile-verified, matrix operand order checked by reading, not run on hardware
//
#include <hip/hip_runtime.h>
#include <hip/hip_bf16.h>
#include <math.h>

// ---------------------------------------------------------------------------
// MambaEncoder for gfx1250 (MI455X). bf16 WMMA GEMMs + f32 VALU elementwise.
// NL=4, D_IN=512, D=768, I=1536, N=16, R=48, K=4, LR=32, B=4, L=1024
// GEMM operands are pre-converted to packed bf16 staging buffers; the WMMA
// inner loop is software-pipelined (double-buffered fragments) so next-step
// b128 loads overlap current-step WMMAs instead of full loadcnt drains.
// All GEMMs are guard-free: K % 32 == 0, N % 16 == 0 (K=48 paths are
// zero-padded to 64 during the bf16 convert; padding is exact).
// ---------------------------------------------------------------------------

#define NL   4
#define D_IN 512
#define DD   768
#define II   1536
#define NN   16
#define RR   48
#define KK   4
#define LRK  32
#define BB   4
#define LL   1024
#define MR   (BB * LL)          // 4096 token rows
#define RPAD 64                 // R=48 padded to 64
#define LSCALE (16.0f / 32.0f)  // SCALE = 16/LR

typedef __attribute__((ext_vector_type(16))) __bf16 v16bf;
typedef __attribute__((ext_vector_type(8)))  __bf16 v8bf;
typedef __attribute__((ext_vector_type(4)))  __bf16 v4bf;
typedef __attribute__((ext_vector_type(8)))  float  v8f;

// Lane fragment: two contiguous 8-element bf16 runs (K = base+0..7, base+16..23),
// each one 16-byte global_load_b128; merged in registers.
static __device__ __forceinline__ v16bf load_frag(const __bf16* __restrict__ p)
{
    const v8bf lo = *(const v8bf*)p;
    const v8bf hi = *(const v8bf*)(p + 16);
    return __builtin_shufflevector(lo, hi, 0, 1, 2, 3, 4, 5, 6, 7,
                                           8, 9, 10, 11, 12, 13, 14, 15);
}

// ---------------------------------------------------------------------------
// C[M x N] = (accum ? C : 0) + alpha * A[M x Kd](lda) * W[N x Kd]^T + bias
// A, W are packed bf16. One wave computes a 16 x (16*NT) strip: the A fragment
// is loaded once per K step and reused across NT WMMAs; fragments are double-
// buffered so loads for step k+1 are in flight while step k's WMMAs execute.
// Requires Kd % 32 == 0, N % (16*NT) == 0, row strides 8-element aligned.
// ---------------------------------------------------------------------------
template <int NT>
__global__ void gemm_wmma_bf16(float* __restrict__ C,
                               const __bf16* __restrict__ A,
                               const __bf16* __restrict__ W,
                               const float* __restrict__ bias,
                               int M, int N, int Kd, int lda,
                               float alpha, int accum)
{
    const int wave    = threadIdx.x >> 5;
    const int lane    = threadIdx.x & 31;
    const int groupsN = (N >> 4) / NT;
    const int tile    = blockIdx.x * 4 + wave;
    const int total   = (M >> 4) * groupsN;
    if (tile >= total) return;                 // wave-uniform: EXEC stays all-ones

    const int tm  = tile / groupsN;
    const int tg  = tile % groupsN;
    const int r16 = lane & 15;
    const int hi  = lane >> 4;                 // K half: 0 -> 0-7/16-23, 1 -> 8-15/24-31

    const __bf16* Arow = A + (size_t)(tm * 16 + r16) * lda + hi * 8;
    const __bf16* Wrow[NT];
    int ncol[NT];
#pragma unroll
    for (int j = 0; j < NT; ++j) {
        ncol[j] = (tg * NT + j) * 16 + r16;
        Wrow[j] = W + (size_t)ncol[j] * Kd + hi * 8;
    }

    v8f acc[NT];
#pragma unroll
    for (int j = 0; j < NT; ++j) acc[j] = v8f{};

    // prologue: load K-step 0 fragments
    v16bf a0 = load_frag(Arow);
    v16bf b0[NT];
#pragma unroll
    for (int j = 0; j < NT; ++j) b0[j] = load_frag(Wrow[j]);

    // pipelined main loop: issue step k+1 loads, then consume step k
    int k0 = 0;
    for (; k0 + 32 < Kd; k0 += 32) {
        __builtin_prefetch(Arow + k0 + 96, 0, 1);          // global_prefetch_b8
        const v16bf a1 = load_frag(Arow + k0 + 32);
        v16bf b1[NT];
#pragma unroll
        for (int j = 0; j < NT; ++j) b1[j] = load_frag(Wrow[j] + k0 + 32);
#pragma unroll
        for (int j = 0; j < NT; ++j)
            acc[j] = __builtin_amdgcn_wmma_f32_16x16x32_bf16(
                         false, a0, false, b0[j], (short)0, acc[j], false, false);
        a0 = a1;
#pragma unroll
        for (int j = 0; j < NT; ++j) b0[j] = b1[j];
    }
    // epilogue: last K step
#pragma unroll
    for (int j = 0; j < NT; ++j)
        acc[j] = __builtin_amdgcn_wmma_f32_16x16x32_bf16(
                     false, a0, false, b0[j], (short)0, acc[j], false, false);

#pragma unroll
    for (int j = 0; j < NT; ++j) {
        const float bv = bias ? bias[ncol[j]] : 0.f;
#pragma unroll
        for (int r = 0; r < 8; ++r) {
            const int m = tm * 16 + r + hi * 8;            // C layout: VGPR r -> M=r+8*hi
            const size_t off = (size_t)m * N + ncol[j];
            const float v = alpha * acc[j][r] + bv;
            C[off] = accum ? (C[off] + v) : v;
        }
    }
}

// dst[r*dld + c .. c+3] = (c < cols) ? bf16(src[r*sld + c .. c+3]) : 0
// Requires sld, dld, cols all multiples of 4 (true at every call site).
__global__ void cvt_bf16_kernel(__bf16* __restrict__ dst, const float* __restrict__ src,
                                int rows, int sld, int dld, int cols)
{
    const size_t q = (size_t)blockIdx.x * blockDim.x + threadIdx.x;
    const size_t idx = q * 4;
    if (idx >= (size_t)rows * dld) return;
    const int r = (int)(idx / dld);
    const int c = (int)(idx % dld);
    v4bf o;
    if (c < cols) {
        const float4 v = *(const float4*)(src + (size_t)r * sld + c);
        o[0] = (__bf16)v.x; o[1] = (__bf16)v.y; o[2] = (__bf16)v.z; o[3] = (__bf16)v.w;
    } else {
        o = v4bf{};
    }
    *(v4bf*)(dst + idx) = o;
}

// ---------------------------------------------------------------------------
// RMS norm per row of D elements: y = x * rsqrt(mean(x^2)+eps) * w
// ---------------------------------------------------------------------------
__global__ void rms_kernel(float* __restrict__ dst, const float* __restrict__ src,
                           const float* __restrict__ w, int D)
{
    const int row = blockIdx.x;
    const float* x = src + (size_t)row * D;
    float*       y = dst + (size_t)row * D;
    float s = 0.f;
    for (int i = threadIdx.x; i < D; i += blockDim.x) { float v = x[i]; s += v * v; }
    __shared__ float red[256];
    red[threadIdx.x] = s;
    __syncthreads();
    for (int st = 128; st > 0; st >>= 1) {
        if (threadIdx.x < st) red[threadIdx.x] += red[threadIdx.x + st];
        __syncthreads();
    }
    const float r = rsqrtf(red[0] / (float)D + 1e-5f);
    for (int i = threadIdx.x; i < D; i += blockDim.x) y[i] = x[i] * r * w[i];
}

// ---------------------------------------------------------------------------
// Depthwise causal conv (K=4) + SiLU. proj rows hold [hs | gate] (stride 2I).
// ---------------------------------------------------------------------------
__global__ void conv_silu_kernel(float* __restrict__ u, const float* __restrict__ proj,
                                 const float* __restrict__ cw, const float* __restrict__ cb)
{
    const size_t idx = (size_t)blockIdx.x * blockDim.x + threadIdx.x;
    if (idx >= (size_t)MR * II) return;
    const int i = (int)(idx % II);
    const size_t bt = idx / II;
    const int t = (int)(bt % LL);
    const int b = (int)(bt / LL);
    float acc = cb[i];
#pragma unroll
    for (int k = 0; k < KK; ++k) {
        const int tt = t - (KK - 1) + k;
        if (tt >= 0)
            acc += proj[((size_t)(b * LL + tt)) * (2 * II) + i] * cw[i * KK + k];
    }
    u[idx] = acc / (1.f + __expf(-acc));       // silu
}

__global__ void softplus_kernel(float* __restrict__ x, size_t n)
{
    const size_t idx = (size_t)blockIdx.x * blockDim.x + threadIdx.x;
    if (idx >= n) return;
    const float v = x[idx];
    x[idx] = (v > 20.f) ? v : log1pf(__expf(v));
}

// ---------------------------------------------------------------------------
// Selective scan: one thread per (b, i) channel, sequential over t, N=16 states.
// B_t = xdb[row*80 + 48 + n], C_t = xdb[row*80 + 64 + n] (16B-aligned float4s).
// ---------------------------------------------------------------------------
__global__ void scan_kernel(float* __restrict__ y, const float* __restrict__ dt,
                            const float* __restrict__ u, const float* __restrict__ xdb,
                            const float* __restrict__ A_log)
{
    const int idx = blockIdx.x * blockDim.x + threadIdx.x;
    if (idx >= BB * II) return;
    const int i = idx % II;
    const int b = idx / II;
    float A[NN], s[NN];
#pragma unroll
    for (int n = 0; n < NN; ++n) { A[n] = -__expf(A_log[i * NN + n]); s[n] = 0.f; }
    for (int t = 0; t < LL; ++t) {
        const size_t row = (size_t)b * LL + t;
        const float dtv = dt[row * II + i];
        const float uv  = u[row * II + i];
        const float du  = dtv * uv;
        const float4* xB = (const float4*)(xdb + row * (RR + 2 * NN) + RR);
        float Bv[NN], Cv[NN];
#pragma unroll
        for (int q = 0; q < 4; ++q) {
            const float4 bb = xB[q], cc = xB[4 + q];
            Bv[4*q+0] = bb.x; Bv[4*q+1] = bb.y; Bv[4*q+2] = bb.z; Bv[4*q+3] = bb.w;
            Cv[4*q+0] = cc.x; Cv[4*q+1] = cc.y; Cv[4*q+2] = cc.z; Cv[4*q+3] = cc.w;
        }
        float yv = 0.f;
#pragma unroll
        for (int n = 0; n < NN; ++n) {
            s[n] = __expf(dtv * A[n]) * s[n] + du * Bv[n];
            yv  += s[n] * Cv[n];
        }
        y[row * II + i] = yv;
    }
}

// y = (y + u * D_skip[i]) * silu(gate);  gate = proj[row*2I + I + i]
__global__ void gate_kernel(float* __restrict__ y, const float* __restrict__ u,
                            const float* __restrict__ proj, const float* __restrict__ Dskip)
{
    const size_t idx = (size_t)blockIdx.x * blockDim.x + threadIdx.x;
    if (idx >= (size_t)MR * II) return;
    const int i = (int)(idx % II);
    const size_t row = idx / II;
    const float g = proj[row * (2 * II) + II + i];
    y[idx] = (y[idx] + u[idx] * Dskip[i]) * (g / (1.f + __expf(-g)));
}

__global__ void add_kernel(float* __restrict__ a, const float* __restrict__ b, size_t n)
{
    const size_t idx = (size_t)blockIdx.x * blockDim.x + threadIdx.x;
    if (idx < n) a[idx] += b[idx];
}

// ---------------------------------------------------------------------------
// Host-side orchestration
// ---------------------------------------------------------------------------
static void launch_gemm(hipStream_t s, float* C, const __bf16* A, const __bf16* W,
                        const float* bias, int M, int N, int Kd, int lda,
                        float alpha, int accum)
{
    const int tilesN = N >> 4;
    if ((tilesN & 3) == 0) {
        const int total = (M >> 4) * (tilesN >> 2);
        gemm_wmma_bf16<4><<<(total + 3) / 4, 128, 0, s>>>(C, A, W, bias, M, N, Kd, lda, alpha, accum);
    } else if ((tilesN & 1) == 0) {
        const int total = (M >> 4) * (tilesN >> 1);
        gemm_wmma_bf16<2><<<(total + 3) / 4, 128, 0, s>>>(C, A, W, bias, M, N, Kd, lda, alpha, accum);
    } else {
        const int total = (M >> 4) * tilesN;
        gemm_wmma_bf16<1><<<(total + 3) / 4, 128, 0, s>>>(C, A, W, bias, M, N, Kd, lda, alpha, accum);
    }
}

static void launch_cvt(hipStream_t s, __bf16* dst, const float* src,
                       int rows, int sld, int dld, int cols)
{
    const size_t n = ((size_t)rows * dld) / 4;   // 4 elements per thread
    cvt_bf16_kernel<<<(int)((n + 255) / 256), 256, 0, s>>>(dst, src, rows, sld, dld, cols);
}

extern "C" void kernel_launch(void* const* d_in, const int* in_sizes, int n_in,
                              void* d_out, int out_size, void* d_ws, size_t ws_size,
                              hipStream_t stream)
{
    const float* x      = (const float*)d_in[0];
    const float* proj_w = (const float*)d_in[1];
    const float* proj_b = (const float*)d_in[2];
    const float* norm_w = (const float*)d_in[3];
    const float* in_w   = (const float*)d_in[4];
    const float* in_la  = (const float*)d_in[5];
    const float* in_lb  = (const float*)d_in[6];
    const float* conv_w = (const float*)d_in[7];
    const float* conv_b = (const float*)d_in[8];
    const float* xp_w   = (const float*)d_in[9];
    const float* xp_la  = (const float*)d_in[10];
    const float* xp_lb  = (const float*)d_in[11];
    const float* dt_w   = (const float*)d_in[12];
    const float* dt_b   = (const float*)d_in[13];
    const float* dt_la  = (const float*)d_in[14];
    const float* dt_lb  = (const float*)d_in[15];
    const float* A_log  = (const float*)d_in[16];
    const float* D_skip = (const float*)d_in[17];
    const float* out_w  = (const float*)d_in[18];
    const float* normf  = (const float*)d_in[19];
    float* out = (float*)d_out;

    // workspace carving: f32 buffers first, then bf16 staging
    float* ws   = (float*)d_ws;
    float* h    = ws;                          // MR * D
    float* hn   = h    + (size_t)MR * DD;      // MR * D
    float* proj = hn   + (size_t)MR * DD;      // MR * 2I
    float* u    = proj + (size_t)MR * 2 * II;  // MR * I
    float* dt   = u    + (size_t)MR * II;      // MR * I
    float* ybuf = dt   + (size_t)MR * II;      // MR * I
    float* yout = ybuf + (size_t)MR * II;      // MR * D
    float* xdb  = yout + (size_t)MR * DD;      // MR * 80
    float* tlo  = xdb  + (size_t)MR * (RR + 2 * NN); // MR * LR
    __bf16* abf = (__bf16*)(tlo + (size_t)MR * LRK); // MR * I   (activation bf16)
    __bf16* wbf = abf + (size_t)MR * II;             // 2I * D   (weight bf16)

    const size_t nTok  = (size_t)MR * II;
    const size_t nTokD = (size_t)MR * DD;
    const int EB = 256;
    const int gTok  = (int)((nTok  + EB - 1) / EB);
    const int gTokD = (int)((nTokD + EB - 1) / EB);

    // h = x @ proj_w^T + proj_b
    launch_cvt(stream, abf, x, MR, D_IN, D_IN, D_IN);
    launch_cvt(stream, wbf, proj_w, DD, D_IN, D_IN, D_IN);
    launch_gemm(stream, h, abf, wbf, proj_b, MR, DD, D_IN, D_IN, 1.f, 0);

    for (int l = 0; l < NL; ++l) {
        const float* l_in_w  = in_w  + (size_t)l * 2 * II * DD;
        const float* l_in_la = in_la + (size_t)l * LRK * DD;
        const float* l_in_lb = in_lb + (size_t)l * 2 * II * LRK;
        const float* l_cw    = conv_w + (size_t)l * II * KK;
        const float* l_cb    = conv_b + (size_t)l * II;
        const float* l_xp_w  = xp_w  + (size_t)l * (RR + 2 * NN) * II;
        const float* l_xp_la = xp_la + (size_t)l * LRK * II;
        const float* l_xp_lb = xp_lb + (size_t)l * (RR + 2 * NN) * LRK;
        const float* l_dt_w  = dt_w  + (size_t)l * II * RR;
        const float* l_dt_b  = dt_b  + (size_t)l * II;
        const float* l_dt_la = dt_la + (size_t)l * LRK * RR;
        const float* l_dt_lb = dt_lb + (size_t)l * II * LRK;
        const float* l_Alog  = A_log + (size_t)l * II * NN;
        const float* l_Dsk   = D_skip + (size_t)l * II;
        const float* l_out_w = out_w + (size_t)l * DD * II;

        // hn = rms(h, norm_w[l]); stage bf16
        rms_kernel<<<MR, 256, 0, stream>>>(hn, h, norm_w + (size_t)l * DD, DD);
        launch_cvt(stream, abf, hn, MR, DD, DD, DD);

        // proj = hn @ in_w^T + SCALE * (hn @ in_la^T) @ in_lb^T
        launch_cvt(stream, wbf, l_in_w, 2 * II, DD, DD, DD);
        launch_gemm(stream, proj, abf, wbf, nullptr, MR, 2 * II, DD, DD, 1.f, 0);
        launch_cvt(stream, wbf, l_in_la, LRK, DD, DD, DD);
        launch_gemm(stream, tlo, abf, wbf, nullptr, MR, LRK, DD, DD, 1.f, 0);
        launch_cvt(stream, abf, tlo, MR, LRK, LRK, LRK);
        launch_cvt(stream, wbf, l_in_lb, 2 * II, LRK, LRK, LRK);
        launch_gemm(stream, proj, abf, wbf, nullptr, MR, 2 * II, LRK, LRK, LSCALE, 1);

        // u = silu(causal_conv(hs) + cb); stage bf16
        conv_silu_kernel<<<gTok, EB, 0, stream>>>(u, proj, l_cw, l_cb);
        launch_cvt(stream, abf, u, MR, II, II, II);

        // xdb = u @ xp_w^T + SCALE * (u @ xp_la^T) @ xp_lb^T
        launch_cvt(stream, wbf, l_xp_w, RR + 2 * NN, II, II, II);
        launch_gemm(stream, xdb, abf, wbf, nullptr, MR, RR + 2 * NN, II, II, 1.f, 0);
        launch_cvt(stream, wbf, l_xp_la, LRK, II, II, II);
        launch_gemm(stream, tlo, abf, wbf, nullptr, MR, LRK, II, II, 1.f, 0);
        launch_cvt(stream, abf, tlo, MR, LRK, LRK, LRK);
        launch_cvt(stream, wbf, l_xp_lb, RR + 2 * NN, LRK, LRK, LRK);
        launch_gemm(stream, xdb, abf, wbf, nullptr, MR, RR + 2 * NN, LRK, LRK, LSCALE, 1);

        // dt = softplus(dt_in @ dt_w^T + SCALE*(dt_in @ dt_la^T) @ dt_lb^T + dt_b)
        // K=48 zero-padded to 64 during bf16 convert (exact).
        launch_cvt(stream, abf, xdb, MR, RR + 2 * NN, RPAD, RR);
        launch_cvt(stream, wbf, l_dt_w, II, RR, RPAD, RR);
        launch_gemm(stream, dt, abf, wbf, l_dt_b, MR, II, RPAD, RPAD, 1.f, 0);
        launch_cvt(stream, wbf, l_dt_la, LRK, RR, RPAD, RR);
        launch_gemm(stream, tlo, abf, wbf, nullptr, MR, LRK, RPAD, RPAD, 1.f, 0);
        launch_cvt(stream, abf, tlo, MR, LRK, LRK, LRK);
        launch_cvt(stream, wbf, l_dt_lb, II, LRK, LRK, LRK);
        launch_gemm(stream, dt, abf, wbf, nullptr, MR, II, LRK, LRK, LSCALE, 1);
        softplus_kernel<<<gTok, EB, 0, stream>>>(dt, nTok);

        // selective scan
        scan_kernel<<<(BB * II + EB - 1) / EB, EB, 0, stream>>>(ybuf, dt, u, xdb, l_Alog);

        // gating: ybuf = (ybuf + u*D_skip) * silu(gate)
        gate_kernel<<<gTok, EB, 0, stream>>>(ybuf, u, proj, l_Dsk);

        // yout = ybuf @ out_w^T ; h += yout
        launch_cvt(stream, abf, ybuf, MR, II, II, II);
        launch_cvt(stream, wbf, l_out_w, DD, II, II, II);
        launch_gemm(stream, yout, abf, wbf, nullptr, MR, DD, II, II, 1.f, 0);
        add_kernel<<<gTokD, EB, 0, stream>>>(h, yout, nTokD);
    }

    // final RMS norm into d_out
    rms_kernel<<<MR, 256, 0, stream>>>(out, h, normf, DD);
}
